// MergeLayer_t_72052371357831
// MI455X (gfx1250) — matrix operand order repaired and checked by reading, not compiled, not run on hardware
//
#include <hip/hip_runtime.h>

#define N_ROWS 8192
#define DIM    4096
#define MBLOCK 128
#define NTOT   128
#define KT     32
#define JSPLIT 8
#define JCHUNK (DIM / JSPLIT)            // 512
#define KSTEPS (DIM / KT)                // 128
#define LDA    40                        // LDS row stride in halves (32 + 8 pad, rows 16B aligned)
#define A1OFF  0
#define A2OFF  (MBLOCK * LDA)            // 5120
#define BWOFF  (2 * MBLOCK * LDA)        // 10240
#define BUFSZ  (3 * MBLOCK * LDA)        // 15360 halves per buffer

typedef _Float16 v8h  __attribute__((ext_vector_type(8)));
typedef _Float16 v16h __attribute__((ext_vector_type(16)));
typedef float    v8f  __attribute__((ext_vector_type(8)));

__device__ __forceinline__ v8f wmma_f16(v16h a, v16h b, v8f c) {
  // D = A(16x32 f16) * B(32x16 f16) + C(16x16 f32)
  return __builtin_amdgcn_wmma_f32_16x16x32_f16(false, a, false, b, (short)0, c,
                                                false, false);
}

// A fragment (16x32, row-major in LDS): lane r = M row; lane-half selects
// K chunks {0..7,16..23} (lanes 0-15) / {8..15,24..31} (lanes 16-31).
__device__ __forceinline__ v16h load_frag_a(const _Float16* tile, int lane) {
  const int r  = lane & 15;
  const int ko = (lane >> 4) << 3;          // 0 or 8
  const _Float16* p = tile + r * LDA + ko;
  union { v16h v; v8h h[2]; } u;
  u.h[0] = *(const v8h*)(p);                // K = ko .. ko+7
  u.h[1] = *(const v8h*)(p + 16);           // K = 16+ko .. 16+ko+7
  return u.v;
}

// B fragment (K=32 x N=16). B rows in LDS are W rows (we need X @ W^T).
// lane r = N column; lane-half selects K {0..7,16..23} / {8..15,24..31}.
__device__ __forceinline__ v16h load_frag_b(const _Float16* tile, int lane) {
  const int r  = lane & 15;
  const int ko = (lane >> 4) << 3;          // 0 or 8
  const _Float16* p = tile + r * LDA + ko;
  union { v16h v; v8h h[2]; } u;
  u.h[0] = *(const v8h*)(p);
  u.h[1] = *(const v8h*)(p + 16);
  return u.v;
}

__device__ __forceinline__ void cvt_store4(_Float16* dst, float4 v) {
  union { _Float16 h[4]; unsigned long long u; } t;
  t.h[0] = (_Float16)v.x; t.h[1] = (_Float16)v.y;
  t.h[2] = (_Float16)v.z; t.h[3] = (_Float16)v.w;
  *(unsigned long long*)dst = t.u;          // ds_store_b64, 8B aligned
}

__global__ void zero_f32(float* __restrict__ p, int n) {
  int i = blockIdx.x * blockDim.x + threadIdx.x;
  if (i < n) p[i] = 0.0f;
}

__global__ void __launch_bounds__(256)
fused_dual_gemm_rowdot(const float* __restrict__ x1,
                       const float* __restrict__ x2,
                       const float* __restrict__ W1,
                       const float* __restrict__ b1,
                       float* __restrict__ out) {
  __shared__ __align__(16) _Float16 smem[2][BUFSZ];   // 61440 B double-buffered

  const int tid  = threadIdx.x;
  const int lane = tid & 31;
  const int wave = tid >> 5;
  const int wm32 = (wave >> 1) * 32;        // wave M offset (0,32,64,96)
  const int wn64 = (wave & 1) * 64;         // wave N offset (0,64)

  const int mBase = blockIdx.x * MBLOCK;
  const int jBase = blockIdx.y * JCHUNK;

  const int c4 = (tid & 7) * 4;             // column (element) within 32-wide panel
  const int r0 = tid >> 3;                  // base row 0..31

  float dacc[2][8] = {};                    // per-row dot partials (2 M tiles)

  float4 ra1[4], ra2[4], rbw[4];            // register stage for next panel

  for (int jt = 0; jt < JCHUNK / NTOT; ++jt) {
    const int jTile = jBase + jt * NTOT;

    v8f c1[2][4] = {};                      // [m-tile][n-tile] accumulators, GEMM1
    v8f c2[2][4] = {};                      // GEMM2

    auto load_panels = [&](int kk) {
      const int kcol = kk * KT + c4;
      #pragma unroll
      for (int q = 0; q < 4; ++q) {
        const int r = r0 + 32 * q;
        ra1[q] = *(const float4*)(x1 + (size_t)(mBase + r) * DIM + kcol);
        ra2[q] = *(const float4*)(x2 + (size_t)(mBase + r) * DIM + kcol);
        rbw[q] = *(const float4*)(W1 + (size_t)(jTile + r) * DIM + kcol);
      }
    };
    auto store_panels = [&](int buf) {
      _Float16* s = &smem[buf][0];
      #pragma unroll
      for (int q = 0; q < 4; ++q) {
        const int r = r0 + 32 * q;
        cvt_store4(s + A1OFF + r * LDA + c4, ra1[q]);
        cvt_store4(s + A2OFF + r * LDA + c4, ra2[q]);
        cvt_store4(s + BWOFF + r * LDA + c4, rbw[q]);
      }
    };

    load_panels(0);
    store_panels(0);
    __syncthreads();

    for (int kk = 0; kk < KSTEPS; ++kk) {
      const int cur  = kk & 1;
      const bool more = (kk + 1) < KSTEPS;
      if (more) load_panels(kk + 1);        // prefetch next K panel into regs

      const _Float16* sb = &smem[cur][0];
      v16h a1f[2], a2f[2];
      #pragma unroll
      for (int rt = 0; rt < 2; ++rt) {
        a1f[rt] = load_frag_a(sb + A1OFF + (wm32 + 16 * rt) * LDA, lane);
        a2f[rt] = load_frag_a(sb + A2OFF + (wm32 + 16 * rt) * LDA, lane);
      }
      #pragma unroll
      for (int nt = 0; nt < 4; ++nt) {
        const v16h bfr = load_frag_b(sb + BWOFF + (wn64 + 16 * nt) * LDA, lane);
        #pragma unroll
        for (int rt = 0; rt < 2; ++rt) {
          c1[rt][nt] = wmma_f16(a1f[rt], bfr, c1[rt][nt]);
          c2[rt][nt] = wmma_f16(a2f[rt], bfr, c2[rt][nt]);
        }
      }

      if (more) store_panels(cur ^ 1);      // fill the other buffer
      __syncthreads();
    }

    // bias + ReLU + elementwise product folded into per-row dot accumulators
    float bvv[4];
    #pragma unroll
    for (int nt = 0; nt < 4; ++nt)
      bvv[nt] = b1[jTile + wn64 + 16 * nt + (lane & 15)];

    #pragma unroll
    for (int rt = 0; rt < 2; ++rt)
      #pragma unroll
      for (int nt = 0; nt < 4; ++nt) {
        const float bv = bvv[nt];
        #pragma unroll
        for (int i = 0; i < 8; ++i) {
          float p = c1[rt][nt][i] + bv; p = p > 0.0f ? p : 0.0f;
          float q = c2[rt][nt][i] + bv; q = q > 0.0f ? q : 0.0f;
          dacc[rt][i] += p * q;
        }
      }
  }

  // Reduce over the 16 N-lanes of each half-wave, then atomically add partials.
  // C layout: VGPR i -> row i (lanes 0-15) / row 8+i (lanes 16-31).
  #pragma unroll
  for (int rt = 0; rt < 2; ++rt) {
    float red[8];
    #pragma unroll
    for (int i = 0; i < 8; ++i) {
      float v = dacc[rt][i];
      v += __shfl_xor(v, 1, 32);
      v += __shfl_xor(v, 2, 32);
      v += __shfl_xor(v, 4, 32);
      v += __shfl_xor(v, 8, 32);
      red[i] = v;
    }
    if ((lane & 15) == 0) {
      const int row = mBase + wm32 + rt * 16 + ((lane >> 4) << 3);
      #pragma unroll
      for (int i = 0; i < 8; ++i)
        atomicAdd(&out[row + i], red[i]);
    }
  }
}

extern "C" void kernel_launch(void* const* d_in, const int* in_sizes, int n_in,
                              void* d_out, int out_size, void* d_ws, size_t ws_size,
                              hipStream_t stream) {
  (void)in_sizes; (void)n_in; (void)out_size; (void)d_ws; (void)ws_size;
  const float* x1 = (const float*)d_in[0];
  const float* x2 = (const float*)d_in[1];
  const float* W1 = (const float*)d_in[2];
  const float* b1 = (const float*)d_in[3];
  float* out = (float*)d_out;

  zero_f32<<<(N_ROWS + 255) / 256, 256, 0, stream>>>(out, N_ROWS);

  dim3 grid(N_ROWS / MBLOCK, JSPLIT);
  fused_dual_gemm_rowdot<<<grid, 256, 0, stream>>>(x1, x2, W1, b1, out);
}
